// TransformerSN_88828513616337
// MI455X (gfx1250) — matrix operand order repaired and checked by reading, not compiled, not run on hardware
//
#include <hip/hip_runtime.h>
#include <hip/hip_bf16.h>
#include <math.h>

typedef _Float16 f16;
typedef _Float16 v16h __attribute__((ext_vector_type(16)));
typedef _Float16 v8h  __attribute__((ext_vector_type(8)));
typedef float    v8f  __attribute__((ext_vector_type(8)));

#define NNODES 65536
#define NGRAPH 512
#define WPB    8                    // waves per block (256 threads)
#define NTILES (NNODES/16)          // 4096 tiles of 16 nodes
#define MAIN_BLOCKS (NTILES/WPB)    // 512

// ---------------------------------------------------------------- utilities
__global__ __launch_bounds__(256) void k_zero(float* __restrict__ p, int n) {
  int i = blockIdx.x * 256 + threadIdx.x;
  if (i < n) p[i] = 0.0f;
}

__global__ __launch_bounds__(256) void k_counts(const int* __restrict__ batch,
                                                float* __restrict__ counts) {
  int e = blockIdx.x * 256 + threadIdx.x;   // exactly NNODES threads
  atomicAdd(&counts[batch[e]], 1.0f);
}

// Pack W2 into the wave32 WMMA B-operand layout, folding 1/sqrt(64*32).
// Bpack[((c*4 + t)*32 + lane)*16 + hh] = B-chunk c (K=c*32..c*32+31), N-tile t.
// lane n<16 holds column n, K = 0..15 of chunk; lane>=16 holds K = 16..31.
// k = m*64 + u maps to W2[m][t*1024 + u*16 + n].
__global__ __launch_bounds__(256) void k_bpack(const float* __restrict__ W2,
                                               f16* __restrict__ Bpack) {
  int id = blockIdx.x * 256 + threadIdx.x;  // 8192 = 64 chunks * 4 tiles * 32 lanes
  int lane = id & 31;
  int ct   = id >> 5;
  int t = ct & 3, c = ct >> 2;
  int n = lane & 15;
  int kbase = c * 32 + (lane >> 4) * 16;
  #pragma unroll
  for (int hh = 0; hh < 16; ++hh) {
    int k = kbase + hh;
    int m = k >> 6, u = k & 63;
    float v = W2[m * 4096 + t * 1024 + u * 16 + n] * 0.02209708691f; // 1/sqrt(2048)
    Bpack[(size_t)id * 16 + hh] = (f16)v;
  }
}

// ---------------------------------------------------------------- main fused kernel
__global__ __launch_bounds__(256) void k_main(
    const float* __restrict__ pos, const int* __restrict__ xid,
    const int* __restrict__ mid, const int* __restrict__ batch,
    const float* __restrict__ emb_z, const float* __restrict__ emb_mol,
    const float* __restrict__ W1, const f16* __restrict__ Bpack,
    float* __restrict__ sums) {
  __shared__ __align__(16) float s_W1[320];            // 10 x 32
  __shared__ __align__(16) f16   s_h[WPB][16][32];     // per-wave h tile (f16)
  __shared__ __align__(16) f16   s_feat[WPB][16][64];  // per-wave feat tile (f16)
  __shared__ __align__(16) float s_sh[WPB][16][16];    // 16 SH values per node
  __shared__ __align__(16) int   s_bid[WPB][16];       // batch id per node

  const int tid  = threadIdx.x;
  const int wave = tid >> 5;
  const int lane = tid & 31;
  const int i    = lane & 15;                 // node within tile
  const int hf   = lane >> 4;                 // lane half
  const int tile = blockIdx.x * WPB + wave;
  const int e    = tile * 16 + i;             // global node id

  // Stage W1 into LDS with the CDNA5 async global->LDS data mover (ASYNCcnt).
  if (wave == 0) {
    #pragma unroll
    for (int j = 0; j < 3; ++j) {
      int idx = j * 32 + lane;                // 16-byte units, 80 total
      if (idx < 80) {
        const float* src = W1 + idx * 4;
        unsigned     dst = (unsigned)(uintptr_t)(void*)&s_W1[idx * 4];
        asm volatile("global_load_async_to_lds_b128 %0, %1, off"
                     :: "v"(dst), "v"(src) : "memory");
      }
    }
    asm volatile("s_wait_asynccnt 0x0" ::: "memory");
  }

  // ---------------- per-node scalar work ----------------
  float px = pos[3 * e], py = pos[3 * e + 1], pz = pos[3 * e + 2];
  float len = sqrtf(px * px + py * py + pz * pz);

  // radial basis: smooth_finite soft-one-hot, cutoff, * sqrt(10); folded const
  float bas[10];
  #pragma unroll
  for (int b = 0; b < 10; ++b) {
    float d  = len * 1.1f - (float)(b + 1);
    float t1 = d + 1.0f, t2 = 1.0f - d;
    float s1 = (t1 > 0.0f) ? __expf(-1.0f / t1) : 0.0f;
    float s2 = (t2 > 0.0f) ? __expf(-1.0f / t2) : 0.0f;
    bas[b] = 26.669302f * (s1 * s2);          // 1.14136*e^2*sqrt(10)
  }

  // feat: u = hf*32 + j  (48 z-embed + 16 mol-embed)
  int xi = xid[e] * 48, mi = mid[e] * 16;
  if (hf == 0) {
    #pragma unroll
    for (int j = 0; j < 32; ++j) s_feat[wave][i][j] = (f16)emb_z[xi + j];
  } else {
    #pragma unroll
    for (int j = 0; j < 16; ++j) s_feat[wave][i][32 + j] = (f16)emb_z[xi + 32 + j];
    #pragma unroll
    for (int j = 0; j < 16; ++j) s_feat[wave][i][48 + j] = (f16)emb_mol[mi + j];
  }
  // spherical harmonics lmax=3 (component normalization), one half-lane per node
  if (hf == 0) {
    float linv = 1.0f / fmaxf(len, 1e-20f);
    float ux = px * linv, uy = py * linv, uz = pz * linv;
    const float S3 = 1.73205081f, S5 = 2.23606798f, S7 = 2.64575131f;
    const float S56 = 0.91287093f, S38 = 0.61237244f;
    float y2 = uy * uy, x2z2 = ux * ux + uz * uz;
    float s20 = S3 * ux * uz, s21 = S3 * ux * uy, s22 = y2 - 0.5f * x2z2;
    float s23 = S3 * uy * uz, s24 = 0.5f * S3 * (uz * uz - ux * ux);
    float* sh = s_sh[wave][i];
    sh[0] = 1.0f;
    sh[1] = S3 * ux; sh[2] = S3 * uy; sh[3] = S3 * uz;
    sh[4] = S5 * s20; sh[5] = S5 * s21; sh[6] = S5 * s22; sh[7] = S5 * s23; sh[8] = S5 * s24;
    float q = 4.0f * y2 - x2z2;
    sh[9]  = S7 * (S56 * (s20 * uz + s24 * ux));
    sh[10] = S7 * (S5 * s20 * uy);
    sh[11] = S7 * (S38 * q * ux);
    sh[12] = S7 * (0.5f * uy * (2.0f * y2 - 3.0f * x2z2));
    sh[13] = S7 * (S38 * uz * q);
    sh[14] = S7 * (S5 * s24 * uy);
    sh[15] = S7 * (S56 * (s24 * uz - s20 * ux));
    s_bid[wave][i] = batch[e];
  }
  __syncthreads();   // W1 (async) + feat/sh staged

  // h[m] for m = hf*16 .. hf*16+15  (normalized SiLU MLP layer); needs s_W1
  #pragma unroll
  for (int mm = 0; mm < 16; ++mm) {
    int m = hf * 16 + mm;
    float s = 0.0f;
    #pragma unroll
    for (int b = 0; b < 10; ++b) s += bas[b] * s_W1[b * 32 + m];
    s *= 0.31622776601f;                      // 1/sqrt(10)
    float hval = 1.67652f * s / (1.0f + __expf(-s));  // SILU_C * silu(s)
    s_h[wave][i][m] = (f16)hval;
  }
  __syncthreads();

  // ---------------- GEMM: xw[16 x 64] = g[16 x 2048] @ W2r via WMMA ----------------
  v8f acc[4];
  #pragma unroll
  for (int t = 0; t < 4; ++t) acc[t] = {};

  // Each lane consumes a fixed 32-element subset of its feat row: hoist to regs.
  const f16* fRow = &s_feat[wave][i][0];
  const v8h fg0 = *(const v8h*)(fRow +      hf * 16);  // u =      hf*16 + 0..7
  const v8h fg1 = *(const v8h*)(fRow +  8 + hf * 16);  // u =  8 + hf*16 + 0..7
  const v8h fg2 = *(const v8h*)(fRow + 32 + hf * 16);  // u = 32 + hf*16 + 0..7
  const v8h fg3 = *(const v8h*)(fRow + 40 + hf * 16);  // u = 40 + hf*16 + 0..7
  const f16* hRow = &s_h[wave][i][0];

  const v16h* bp = ((const v16h*)Bpack) + lane;  // per-lane base; tile stride 32, chunk stride 128

  // Register ping-pong on B: chunk 2m in b0 / 2m+1 in b1, prefetch one chunk ahead.
  v16h b0[4], b1[4];
  #pragma unroll
  for (int t = 0; t < 4; ++t) b0[t] = bp[t * 32];          // chunk 0

  f16 hv = hRow[0];
  for (int m = 0; m < 32; ++m) {
    // prefetch B for chunk 2m+1
    {
      const v16h* bn = bp + (size_t)(2 * m + 1) * 128;
      #pragma unroll
      for (int t = 0; t < 4; ++t) b1[t] = bn[t * 32];
    }
    f16 hvn = hRow[(m < 31) ? (m + 1) : m];                // prefetch next h

    // chunk 2m (uoff = 0)
    v8h alo0 = fg0 * hv, ahi0 = fg1 * hv;
    v16h a0 = __builtin_shufflevector(alo0, ahi0,
        0, 1, 2, 3, 4, 5, 6, 7, 8, 9, 10, 11, 12, 13, 14, 15);
    #pragma unroll
    for (int t = 0; t < 4; ++t)
      acc[t] = __builtin_amdgcn_wmma_f32_16x16x32_f16(
          false, a0, false, b0[t], (short)0, acc[t], false, false);

    // prefetch B for chunk 2m+2
    if (m < 31) {
      const v16h* bn = bp + (size_t)(2 * m + 2) * 128;
      #pragma unroll
      for (int t = 0; t < 4; ++t) b0[t] = bn[t * 32];
    }

    // chunk 2m+1 (uoff = 32)
    v8h alo1 = fg2 * hv, ahi1 = fg3 * hv;
    v16h a1 = __builtin_shufflevector(alo1, ahi1,
        0, 1, 2, 3, 4, 5, 6, 7, 8, 9, 10, 11, 12, 13, 14, 15);
    #pragma unroll
    for (int t = 0; t < 4; ++t)
      acc[t] = __builtin_amdgcn_wmma_f32_16x16x32_f16(
          false, a1, false, b1[t], (short)0, acc[t], false, false);

    hv = hvn;
  }

  // ---------------- epilogue: xw * SH, scatter-add into segment sums ----------------
  // D layout: lane n<16 -> rows 0..7 (f[j]=D[j,n]); lane>=16 -> rows 8..15, col n-16.
  const int w = lane & 15;
  const int lOff[4] = {0, 16, 64, 144};
  const int sOff[4] = {0, 1, 4, 9};
  #pragma unroll
  for (int t = 0; t < 4; ++t) {
    int nc = 2 * t + 1;
    #pragma unroll
    for (int j = 0; j < 8; ++j) {
      int node = hf * 8 + j;
      float xw = acc[t][j];
      int g = s_bid[wave][node];
      float* dst = sums + (size_t)g * 256 + lOff[t] + w * nc;
      const float* shn = &s_sh[wave][node][sOff[t]];
      for (int cc = 0; cc < nc; ++cc) atomicAdd(dst + cc, xw * shn[cc]);
    }
  }
}

// ---------------------------------------------------------------- finalize mean
__global__ __launch_bounds__(256) void k_final(const float* __restrict__ sums,
                                               const float* __restrict__ counts,
                                               float* __restrict__ out) {
  int id = blockIdx.x * 256 + threadIdx.x;  // 512*256 threads exactly
  int g = id >> 8;
  out[id] = sums[id] / fmaxf(counts[g], 1.0f);
}

// ---------------------------------------------------------------- host entry
extern "C" void kernel_launch(void* const* d_in, const int* in_sizes, int n_in,
                              void* d_out, int out_size, void* d_ws, size_t ws_size,
                              hipStream_t stream) {
  (void)in_sizes; (void)n_in; (void)out_size; (void)ws_size;
  const float* pos     = (const float*)d_in[0];
  const int*   xid     = (const int*)d_in[1];
  const int*   mid     = (const int*)d_in[2];
  const int*   batch   = (const int*)d_in[3];
  const float* emb_z   = (const float*)d_in[4];
  const float* emb_mol = (const float*)d_in[5];
  const float* W1      = (const float*)d_in[6];
  const float* W2      = (const float*)d_in[7];
  float* out = (float*)d_out;

  char*  ws     = (char*)d_ws;
  float* sums   = (float*)ws;                              // 512*256 f32
  float* counts = (float*)(ws + (size_t)NGRAPH * 256 * 4); // 512 f32
  f16*   Bpack  = (f16*)(ws + (size_t)NGRAPH * 256 * 4 + 2048); // 128K f16 (32B aligned)

  const int nzero = NGRAPH * 256 + NGRAPH;                 // sums + counts contiguous
  k_zero  <<<(nzero + 255) / 256, 256, 0, stream>>>((float*)ws, nzero);
  k_bpack <<<32, 256, 0, stream>>>(W2, Bpack);
  k_counts<<<NNODES / 256, 256, 0, stream>>>(batch, counts);
  k_main  <<<MAIN_BLOCKS, 256, 0, stream>>>(pos, xid, mid, batch, emb_z, emb_mol,
                                            W1, Bpack, sums);
  k_final <<<NGRAPH, 256, 0, stream>>>(sums, counts, out);
}
